// CausalSpatialAttention_39891656245439
// MI455X (gfx1250) — matrix operand order repaired and checked by reading, not compiled, not run on hardware
//
#include <hip/hip_runtime.h>

// ---------------------------------------------------------------------------
// CausalSpatialAttention for MI455X (gfx1250, wave32, WMMA 16x16x32 bf16)
// B*T=32, N=512, D=256, H=8, hd=32  (hd == WMMA K depth for bf16)
// Flash-style fused softmax: the 256MB score tensor never touches HBM.
// v2: 16x64 GEMM wave-tiles (4x A-fragment reuse), hoisted A_causal pointers,
//     global_prefetch of next K/V block, vectorized bf16 converts.
// ---------------------------------------------------------------------------

typedef __attribute__((ext_vector_type(16))) __bf16 v16bf;
typedef __attribute__((ext_vector_type(8)))  __bf16 v8bf;
typedef __attribute__((ext_vector_type(8)))  float  v8f;
typedef __attribute__((ext_vector_type(4)))  float  v4f;

#define BT   32
#define NSP  512
#define DMODEL 256
#define NH   8
#define HD   32
#define TFRAMES 8
#define QK_SCALE 0.17677669529663687f  /* 32^-0.5 */

__device__ __forceinline__ __bf16 f2bf(float f) {
    union { float f; unsigned u; } a; a.f = f;
    unsigned r = a.u + 0x7FFFu + ((a.u >> 16) & 1u);   // round-to-nearest-even
    unsigned short h = (unsigned short)(r >> 16);
    return __builtin_bit_cast(__bf16, h);
}

__device__ __forceinline__ v8f wmma_bf16(v16bf a, v16bf b, v8f c) {
    // emits v_wmma_f32_16x16x32_bf16
    return __builtin_amdgcn_wmma_f32_16x16x32_bf16(
        /*neg_a=*/false, a, /*neg_b=*/false, b,
        /*c_mod=*/(short)0, c, /*reuse_a=*/false, /*reuse_b=*/false);
}

// A-fragment loader: 16x32 bf16 tile, row = lane%16.
// Lanes 0-15 hold K 0..15 (elems 0-15);
// lanes 16-31 hold K 16..31 (contiguous, matching B loads).
__device__ __forceinline__ v16bf load_a16(const __bf16* row, int hl) {
    v8bf lo = *(const v8bf*)(row + hl * 16);
    v8bf hi = *(const v8bf*)(row + hl * 16 + 8);
    v16bf r;
#pragma unroll
    for (int i = 0; i < 8; ++i) { r[i] = lo[i]; r[i + 8] = hi[i]; }
    return r;
}

// --------------------- f32 -> bf16 convert (8 elems/thread) ----------------
__global__ void cvt_bf16_kernel(const float* __restrict__ in,
                                __bf16* __restrict__ out, int n8) {
    int i = blockIdx.x * blockDim.x + threadIdx.x;
    if (i < n8) {
        v4f a = ((const v4f*)in)[2 * i];
        v4f b = ((const v4f*)in)[2 * i + 1];
        v8bf r;
#pragma unroll
        for (int j = 0; j < 4; ++j) { r[j] = f2bf(a[j]); r[4 + j] = f2bf(b[j]); }
        ((v8bf*)out)[i] = r;
    }
}

// --------------------------- QKV projection GEMM ---------------------------
// C[16384,768] = x[16384,256] @ Wqkv^T.  Wave computes a 16x64 tile:
// one A-fragment feeds 4 WMMAs per K-step.  Results scattered into
//   q[bt,h,n,hd] * scale, k[bt,h,n,hd], vT[bt,h,hd,n]   (all bf16)
__global__ __launch_bounds__(256) void qkv_gemm_kernel(
    const __bf16* __restrict__ xb, const __bf16* __restrict__ wb,
    __bf16* __restrict__ qb, __bf16* __restrict__ kb, __bf16* __restrict__ vtb) {
    int wave = threadIdx.x >> 5, lane = threadIdx.x & 31;
    int hl = lane >> 4, ln = lane & 15;
    int job = blockIdx.x * 8 + wave;             // 1024 m-tiles x 12 n-supertiles
    int jm = job / 12, jn = job - jm * 12;
    int m0 = jm << 4, n0 = jn << 6;

    const __bf16* arow = xb + (size_t)(m0 + ln) * DMODEL;
    const __bf16* br0  = wb + (size_t)(n0 + ln) * DMODEL;
    const __bf16* br1  = br0 + 16 * DMODEL;
    const __bf16* br2  = br0 + 32 * DMODEL;
    const __bf16* br3  = br0 + 48 * DMODEL;

    v8f ac0 = {}, ac1 = {}, ac2 = {}, ac3 = {};
#pragma unroll
    for (int k0 = 0; k0 < DMODEL; k0 += 32) {
        v16bf a  = load_a16(arow + k0, hl);
        v16bf b0 = *(const v16bf*)(br0 + k0 + hl * 16);
        v16bf b1 = *(const v16bf*)(br1 + k0 + hl * 16);
        v16bf b2 = *(const v16bf*)(br2 + k0 + hl * 16);
        v16bf b3 = *(const v16bf*)(br3 + k0 + hl * 16);
        ac0 = wmma_bf16(a, b0, ac0);
        ac1 = wmma_bf16(a, b1, ac1);
        ac2 = wmma_bf16(a, b2, ac2);
        ac3 = wmma_bf16(a, b3, ac3);
    }

    v8f accs[4] = {ac0, ac1, ac2, ac3};
#pragma unroll
    for (int t = 0; t < 4; ++t) {
#pragma unroll
        for (int i = 0; i < 8; ++i) {
            int m = m0 + i + (hl << 3);          // row in [0,16384)
            int n = n0 + t * 16 + ln;            // col in [0,768)
            int bt = m >> 9, sp = m & (NSP - 1);
            int which = n >> 8, hh = (n >> 5) & 7, d = n & (HD - 1);
            size_t bh = (size_t)(bt * NH + hh);
            float v = accs[t][i];
            if (which == 0)
                qb[((bh * NSP + sp) << 5) + d] = f2bf(v * QK_SCALE);
            else if (which == 1)
                kb[((bh * NSP + sp) << 5) + d] = f2bf(v);
            else
                vtb[((bh << 5) + d) * NSP + sp] = f2bf(v);
        }
    }
}

// ------------------------ fused flash attention ----------------------------
// grid = (4, H, BT); block = 256 (8 waves); wave owns 16 q rows.
__global__ __launch_bounds__(256) void attn_kernel(
    const __bf16* __restrict__ qb, const __bf16* __restrict__ kb,
    const __bf16* __restrict__ vtb, const float* __restrict__ Acausal,
    const float* __restrict__ lamp, __bf16* __restrict__ att_out) {
    __shared__ __align__(32) __bf16 plds[8][16 * 32];   // per-wave P staging

    int wave = threadIdx.x >> 5, lane = threadIdx.x & 31;
    int hl = lane >> 4, ln = lane & 15;
    int bt = blockIdx.z, h = blockIdx.y;
    int q0 = blockIdx.x * 128 + wave * 16;
    int bC = bt / TFRAMES;                       // jnp.repeat over frames
    float lam = lamp[0];

    size_t bh = (size_t)(bt * NH + h) * (NSP * HD);
    const __bf16* qptr = qb + bh;
    const __bf16* kptr = kb + bh;
    const __bf16* vptr = vtb + bh;
    __bf16* pl = &plds[wave][0];

    v16bf aq = load_a16(qptr + (size_t)(q0 + ln) * HD, hl);

    // hoisted per-row A_causal pointers (8 rows per lane)
    const float* Arp[8];
#pragma unroll
    for (int i = 0; i < 8; ++i)
        Arp[i] = Acausal + ((size_t)bC * NSP + (q0 + i + (hl << 3))) * NSP;

    v8f o0 = {}, o1 = {};
    float mrow[8], lrow[8];
#pragma unroll
    for (int i = 0; i < 8; ++i) { mrow[i] = -1e30f; lrow[i] = 0.f; }

    for (int k0 = 0; k0 < NSP; k0 += 32) {
        // prefetch next K/V block into WGP$ (global_prefetch_b8)
        if (k0 + 32 < NSP) {
            __builtin_prefetch(kptr + (size_t)(k0 + 32 + ln) * HD, 0, 3);
            __builtin_prefetch(vptr + (size_t)ln * NSP + k0 + 32, 0, 3);
        }

        // S = (q*scale) @ k^T : two 16x16 tiles, K = hd = 32 (one WMMA each)
        v16bf b0 = *(const v16bf*)(kptr + (size_t)(k0 + ln) * HD + hl * 16);
        v16bf b1 = *(const v16bf*)(kptr + (size_t)(k0 + 16 + ln) * HD + hl * 16);
        v8f s0 = {}, s1 = {};
        s0 = wmma_bf16(aq, b0, s0);
        s1 = wmma_bf16(aq, b1, s1);

        // + lambda * A_causal, then online softmax update (rows across 16 lanes)
#pragma unroll
        for (int i = 0; i < 8; ++i) {
            const float* Ar = Arp[i] + k0;
            float t0 = s0[i] + lam * Ar[ln];
            float t1 = s1[i] + lam * Ar[16 + ln];
            float mx = fmaxf(t0, t1);
#pragma unroll
            for (int msk = 1; msk < 16; msk <<= 1)
                mx = fmaxf(mx, __shfl_xor(mx, msk, 32));
            float mnew = fmaxf(mrow[i], mx);
            float al = __expf(mrow[i] - mnew);
            float p0 = __expf(t0 - mnew);
            float p1 = __expf(t1 - mnew);
            float rs = p0 + p1;
#pragma unroll
            for (int msk = 1; msk < 16; msk <<= 1)
                rs += __shfl_xor(rs, msk, 32);
            lrow[i] = lrow[i] * al + rs;
            mrow[i] = mnew;
            o0[i] *= al; o1[i] *= al;
            s0[i] = p0;  s1[i] = p1;
        }

        // P (C-layout f32) -> bf16 A-layout via wave-private LDS tile
#pragma unroll
        for (int i = 0; i < 8; ++i) {
            int r = i + (hl << 3);
            pl[r * 32 + ln]      = f2bf(s0[i]);
            pl[r * 32 + 16 + ln] = f2bf(s1[i]);
        }
        __syncthreads();   // uniform; orders same-wave LDS RAW at IR level

        v16bf ap  = load_a16(pl + ln * 32, hl);
        v16bf bv0 = *(const v16bf*)(vptr + (size_t)ln * NSP + k0 + hl * 16);
        v16bf bv1 = *(const v16bf*)(vptr + (size_t)(ln + 16) * NSP + k0 + hl * 16);
        o0 = wmma_bf16(ap, bv0, o0);       // O[:, 0:16]  += P @ V
        o1 = wmma_bf16(ap, bv1, o1);       // O[:, 16:32] += P @ V
    }

    // normalize and emit bf16 in [BT*N, D] layout (head-transposed already)
#pragma unroll
    for (int i = 0; i < 8; ++i) {
        float inv = 1.0f / lrow[i];
        int row = q0 + i + (hl << 3);
        size_t ob = ((size_t)bt * NSP + row) * DMODEL + h * HD;
        att_out[ob + ln]      = f2bf(o0[i] * inv);
        att_out[ob + 16 + ln] = f2bf(o1[i] * inv);
    }
}

// ------------------------- output projection GEMM --------------------------
// Y[16384,256] = att(bf16) @ Wproj^T + b.  16x64 wave tiles.
__global__ __launch_bounds__(256) void proj_gemm_kernel(
    const __bf16* __restrict__ ab, const __bf16* __restrict__ wb,
    const float* __restrict__ bias, float* __restrict__ out) {
    int wave = threadIdx.x >> 5, lane = threadIdx.x & 31;
    int hl = lane >> 4, ln = lane & 15;
    int job = blockIdx.x * 8 + wave;             // 1024 m-tiles x 4 n-supertiles
    int jm = job >> 2, jn = job & 3;
    int m0 = jm << 4, n0 = jn << 6;

    const __bf16* arow = ab + (size_t)(m0 + ln) * DMODEL;
    const __bf16* br0  = wb + (size_t)(n0 + ln) * DMODEL;
    const __bf16* br1  = br0 + 16 * DMODEL;
    const __bf16* br2  = br0 + 32 * DMODEL;
    const __bf16* br3  = br0 + 48 * DMODEL;

    v8f ac0 = {}, ac1 = {}, ac2 = {}, ac3 = {};
#pragma unroll
    for (int k0 = 0; k0 < DMODEL; k0 += 32) {
        v16bf a  = load_a16(arow + k0, hl);
        v16bf b0 = *(const v16bf*)(br0 + k0 + hl * 16);
        v16bf b1 = *(const v16bf*)(br1 + k0 + hl * 16);
        v16bf b2 = *(const v16bf*)(br2 + k0 + hl * 16);
        v16bf b3 = *(const v16bf*)(br3 + k0 + hl * 16);
        ac0 = wmma_bf16(a, b0, ac0);
        ac1 = wmma_bf16(a, b1, ac1);
        ac2 = wmma_bf16(a, b2, ac2);
        ac3 = wmma_bf16(a, b3, ac3);
    }

    v8f accs[4] = {ac0, ac1, ac2, ac3};
#pragma unroll
    for (int t = 0; t < 4; ++t) {
        float bv = bias[n0 + t * 16 + ln];
#pragma unroll
        for (int i = 0; i < 8; ++i) {
            int m = m0 + i + (hl << 3);
            out[(size_t)m * DMODEL + n0 + t * 16 + ln] = accs[t][i] + bv;
        }
    }
}

// ---------------------------------------------------------------------------
extern "C" void kernel_launch(void* const* d_in, const int* in_sizes, int n_in,
                              void* d_out, int out_size, void* d_ws, size_t ws_size,
                              hipStream_t stream) {
    const float* x     = (const float*)d_in[0];   // [32*512, 256]
    const float* A     = (const float*)d_in[1];   // [4, 512, 512]
    const float* Wqkv  = (const float*)d_in[2];   // [768, 256]
    const float* Wproj = (const float*)d_in[3];   // [256, 256]
    const float* bproj = (const float*)d_in[4];   // [256]
    const float* lam   = (const float*)d_in[5];   // [1]
    (void)in_sizes; (void)n_in; (void)out_size; (void)ws_size;

    char* ws = (char*)d_ws;
    size_t off = 0;
    auto alloc = [&](size_t bytes) -> void* {
        void* p = ws + off; off += (bytes + 255) & ~(size_t)255; return p;
    };
    const size_t NX = (size_t)BT * NSP * DMODEL;        // 4,194,304
    __bf16* xb     = (__bf16*)alloc(NX * 2);
    __bf16* wqkvb  = (__bf16*)alloc((size_t)3 * DMODEL * DMODEL * 2);
    __bf16* wprojb = (__bf16*)alloc((size_t)DMODEL * DMODEL * 2);
    __bf16* qb     = (__bf16*)alloc(NX * 2);
    __bf16* kb     = (__bf16*)alloc(NX * 2);
    __bf16* vtb    = (__bf16*)alloc(NX * 2);
    __bf16* attb   = (__bf16*)alloc(NX * 2);
    // total workspace ~ 43 MB

    // 1) bf16 converts (8 elements per thread, b128 traffic)
    cvt_bf16_kernel<<<(int)((NX / 8 + 255) / 256), 256, 0, stream>>>(
        x, xb, (int)(NX / 8));
    cvt_bf16_kernel<<<(3 * DMODEL * DMODEL / 8 + 255) / 256, 256, 0, stream>>>(
        Wqkv, wqkvb, 3 * DMODEL * DMODEL / 8);
    cvt_bf16_kernel<<<(DMODEL * DMODEL / 8 + 255) / 256, 256, 0, stream>>>(
        Wproj, wprojb, DMODEL * DMODEL / 8);

    // 2) QKV projection: 1024 x 12 wave-jobs (16x64 tiles), 8 per block
    qkv_gemm_kernel<<<1536, 256, 0, stream>>>(xb, wqkvb, qb, kb, vtb);

    // 3) fused flash attention: (q-block, head, batch*frame)
    dim3 ag(NSP / 128, NH, BT);
    attn_kernel<<<ag, 256, 0, stream>>>(qb, kb, vtb, A, lam, attb);

    // 4) output projection: 1024 x 4 wave-jobs (16x64 tiles), 8 per block
    proj_gemm_kernel<<<512, 256, 0, stream>>>(attb, wprojb, bproj, (float*)d_out);
}
